// PatchQuantumGenerator_86577950753233
// MI455X (gfx1250) — matrix operand order repaired and checked by reading, not compiled, not run on hardware
//
#include <hip/hip_runtime.h>
#include <hip/hip_bf16.h>

typedef __attribute__((ext_vector_type(16))) _Float16 v16h;
typedef __attribute__((ext_vector_type(8)))  float    v8f;

#define C1 128
#define C2 64
#define TLEN 32
#define KTOT 384   // 3*128, ordered dk*128+ci

// ---------------------------------------------------------------------------
// ws float layout:
//   images[B*32]
//   stat[768]: [0..127]=sum1 [128..255]=sq1 [256..383]=sc1 [384..511]=sh1
//              [512..575]=sum2 [576..639]=sq2 [640..703]=sc2 [704..767]=sh2
//   Bpanel: _Float16[64*384]  (W2 effective weights, [co][dk*128+ci])
// ---------------------------------------------------------------------------

__global__ void k_init(const float* __restrict__ w2,
                       float* __restrict__ stat,
                       _Float16* __restrict__ Bp)
{
    int tid = threadIdx.x;
    for (int i = tid; i < 768; i += 256) stat[i] = 0.0f;
    // effective conv2 weight: y2[co,t] = sum_{ci,dk} h1n[ci, t+dk-1] * w2[ci, co, 2-dk]
    for (int i = tid; i < C2 * KTOT; i += 256) {
        int co = i / KTOT;
        int kk = i % KTOT;
        int dk = kk >> 7;
        int ci = kk & 127;
        Bp[i] = (_Float16)w2[ci * (C2 * 3) + co * 3 + (2 - dk)];
    }
}

// ---------------------------------------------------------------------------
__global__ void k_patches(const float* __restrict__ x,
                          const float* __restrict__ qp,
                          float* __restrict__ images, int B)
{
    int b = blockIdx.x * blockDim.x + threadIdx.x;
    if (b >= B) return;
    float xr[4];
#pragma unroll
    for (int q = 0; q < 4; ++q) xr[q] = x[b * 4 + q];
#pragma unroll
    for (int g = 0; g < 4; ++g) {
        float cs[4], sn[4];
#pragma unroll
        for (int q = 0; q < 4; ++q) {
            float th = 0.5f * (xr[q] + qp[g * 4 + q]);
            cs[q] = cosf(th);
            sn[q] = sinf(th);
        }
        float p[8];
        float tot = 0.0f;
#pragma unroll
        for (int i = 0; i < 16; ++i) {
            float a = 1.0f;
#pragma unroll
            for (int q = 0; q < 4; ++q)
                a *= ((i >> (3 - q)) & 1) ? sn[q] : cs[q];
            float pr = a * a;        // sign^2 == 1
            tot += pr;
            if (i < 8) p[i] = pr;
        }
        float inv = 1.0f / tot;
        float mx = 0.0f;
#pragma unroll
        for (int i = 0; i < 8; ++i) { p[i] *= inv; mx = fmaxf(mx, p[i]); }
        float im = 1.0f / mx;
#pragma unroll
        for (int i = 0; i < 8; ++i) images[b * 32 + g * 8 + i] = p[i] * im;
    }
}

// ---------------------------------------------------------------------------
// BN1 statistics: recompute relu(conv1) over the batch; 16 channels per block.x
__global__ void k_stats1(const float* __restrict__ images,
                         const float* __restrict__ w1, const float* __restrict__ b1,
                         float* __restrict__ gsum, float* __restrict__ gsq,
                         int chunk)
{
    __shared__ float sSum[16], sSq[16];
    const int tid = threadIdx.x;
    const int cbase = blockIdx.x * 16;
    if (tid < 16) { sSum[tid] = 0.0f; sSq[tid] = 0.0f; }
    __syncthreads();

    float wA[16], wB[16], wC[16], bb[16];
#pragma unroll
    for (int c = 0; c < 16; ++c) {
        wA[c] = w1[(cbase + c) * 3 + 2];   // tap x[t-1]
        wB[c] = w1[(cbase + c) * 3 + 1];   // tap x[t]
        wC[c] = w1[(cbase + c) * 3 + 0];   // tap x[t+1]
        bb[c] = b1[cbase + c];
    }
    float sum[16] = {}, sq[16] = {};
    const int b0 = blockIdx.y * chunk;
    for (int b = b0 + tid; b < b0 + chunk; b += blockDim.x) {
        const float* im = images + (size_t)b * 32;
        float i0 = 0.0f, i1 = im[0];
        for (int t = 0; t < TLEN; ++t) {
            float i2 = (t < TLEN - 1) ? im[t + 1] : 0.0f;
#pragma unroll
            for (int c = 0; c < 16; ++c) {
                float v = fmaxf(fmaf(i0, wA[c], fmaf(i1, wB[c], fmaf(i2, wC[c], bb[c]))), 0.0f);
                sum[c] += v;
                sq[c] = fmaf(v, v, sq[c]);
            }
            i0 = i1; i1 = i2;
        }
    }
#pragma unroll
    for (int c = 0; c < 16; ++c) { atomicAdd(&sSum[c], sum[c]); atomicAdd(&sSq[c], sq[c]); }
    __syncthreads();
    if (tid < 16) {
        atomicAdd(&gsum[cbase + tid], sSum[tid]);
        atomicAdd(&gsq[cbase + tid], sSq[tid]);
    }
}

// ---------------------------------------------------------------------------
__global__ void k_finalize(const float* __restrict__ gs, const float* __restrict__ gq,
                           const float* __restrict__ gamma, const float* __restrict__ beta,
                           float* __restrict__ sc, float* __restrict__ sh,
                           int nch, float invN)
{
    int c = threadIdx.x;
    if (c < nch) {
        float mean = gs[c] * invN;
        float var  = gq[c] * invN - mean * mean;
        float s = gamma[c] * rsqrtf(var + 1e-5f);
        sc[c] = s;
        sh[c] = beta[c] - mean * s;
    }
}

// ---------------------------------------------------------------------------
// Main WMMA kernel. 128 threads = 4 waves; wave w owns sample blockIdx.x*4+w.
// Both 16-row M-tiles are carried through one K-loop (8 accumulators) so each
// B fragment load feeds two v_wmma ops (DS bandwidth is the pacing resource).
// MODE 0: accumulate BN2 statistics of relu(conv2+b2).
// MODE 1: apply BN2, then conv3 + tanh -> out.
template <int MODE>
__global__ void __launch_bounds__(128)
k_conv2(const float* __restrict__ images,
        const float* __restrict__ w1, const float* __restrict__ b1,
        const float* __restrict__ sc1, const float* __restrict__ sh1,
        const _Float16* __restrict__ Bp,
        const float* __restrict__ b2,
        const float* __restrict__ sc2, const float* __restrict__ sh2,
        const float* __restrict__ w3, const float* __restrict__ b3,
        float* __restrict__ gsum2, float* __restrict__ gsq2,
        float* __restrict__ out)
{
    __shared__ __align__(16) _Float16 sB[C2 * KTOT];              // 48 KB
    __shared__ __align__(16) _Float16 sH1[4][34 * C1];            // 34 KB (padded rows t=-1,32)
    __shared__ float sImg[4][34];
    __shared__ float sStat[(MODE == 0) ? 2 * C2 : 1];
    __shared__ __align__(16) _Float16 sH2[(MODE == 1) ? 4 : 1][(MODE == 1) ? 34 * C2 : 1];

    const int tid  = threadIdx.x;
    const int w    = tid >> 5;       // wave id 0..3
    const int lane = tid & 31;
    const int half = lane >> 4;
    const int lm   = lane & 15;
    const int b    = blockIdx.x * 4 + w;

    // cooperative fill of B panel (16B chunks)
    for (int i = tid; i < (C2 * KTOT) / 8; i += blockDim.x)
        ((int4*)sB)[i] = ((const int4*)Bp)[i];

    if (MODE == 0) {
        if (tid < 2 * C2) sStat[tid] = 0.0f;
    }

    // per-wave padded image row
    sImg[w][lane + 1] = images[(size_t)b * TLEN + lane];
    if (lane == 0) { sImg[w][0] = 0.0f; sImg[w][33] = 0.0f; }

    // per-wave h1 panel: relu(conv1+b1)*sc1+sh1, f16, rows padded with zeros
    for (int c = lane; c < C1; c += 32) {
        float wA = w1[c * 3 + 2], wB = w1[c * 3 + 1], wC = w1[c * 3 + 0];
        float bc = b1[c], sc = sc1[c], sh = sh1[c];
        sH1[w][c] = (_Float16)0.0f;
        sH1[w][33 * C1 + c] = (_Float16)0.0f;
        for (int t = 0; t < TLEN; ++t) {
            float v = fmaxf(fmaf(sImg[w][t], wA,
                           fmaf(sImg[w][t + 1], wB,
                           fmaf(sImg[w][t + 2], wC, bc))), 0.0f);
            sH1[w][(t + 1) * C1 + c] = (_Float16)fmaf(v, sc, sh);
        }
    }
    __syncthreads();   // sB visible to all waves

    // K-loop over 12 steps of 32; 8 live accumulators (2 M-tiles x 4 N-tiles)
    v8f acc[2][4] = {};
    for (int ks = 0; ks < 12; ++ks) {
        const int kb0 = ks * 32 + half * 8;
        const int kb1 = kb0 + 16;
        v16h a[2];
#pragma unroll
        for (int mt = 0; mt < 2; ++mt) {
            const int t0 = mt * 16;
            const _Float16* pa0 = &sH1[w][(t0 + lm + (kb0 >> 7)) * C1 + (kb0 & 127)];
            const _Float16* pa1 = &sH1[w][(t0 + lm + (kb1 >> 7)) * C1 + (kb1 & 127)];
#pragma unroll
            for (int j = 0; j < 8; ++j) { a[mt][j] = pa0[j]; a[mt][j + 8] = pa1[j]; }
        }
#pragma unroll
        for (int nt = 0; nt < 4; ++nt) {
            const _Float16* pb = &sB[(nt * 16 + lm) * KTOT + ks * 32 + half * 8];
            v16h bf;
#pragma unroll
            for (int j = 0; j < 8; ++j) { bf[j] = pb[j]; bf[j + 8] = pb[j + 16]; }
            acc[0][nt] = __builtin_amdgcn_wmma_f32_16x16x32_f16(
                false, a[0], false, bf, (short)0, acc[0][nt], false, false);
            acc[1][nt] = __builtin_amdgcn_wmma_f32_16x16x32_f16(
                false, a[1], false, bf, (short)0, acc[1][nt], false, false);
        }
    }

    // epilogue: lane holds co = nt*16+lm, rows t = mt*16 + half*8 + r
#pragma unroll
    for (int mt = 0; mt < 2; ++mt) {
        const int t0 = mt * 16;
#pragma unroll
        for (int nt = 0; nt < 4; ++nt) {
            const int co = nt * 16 + lm;
            const float bb = b2[co];
            if (MODE == 0) {
                float s = 0.0f, q = 0.0f;
#pragma unroll
                for (int r = 0; r < 8; ++r) {
                    float v = fmaxf(acc[mt][nt][r] + bb, 0.0f);
                    s += v;
                    q = fmaf(v, v, q);
                }
                atomicAdd(&sStat[co], s);
                atomicAdd(&sStat[C2 + co], q);
            } else {
                const float sc = sc2[co], sh = sh2[co];
#pragma unroll
                for (int r = 0; r < 8; ++r) {
                    float v = fmaxf(acc[mt][nt][r] + bb, 0.0f);
                    int t = t0 + half * 8 + r;
                    sH2[w][(t + 1) * C2 + co] = (_Float16)fmaf(v, sc, sh);
                }
            }
        }
    }

    if (MODE == 0) {
        __syncthreads();
        if (tid < C2) {
            atomicAdd(&gsum2[tid], sStat[tid]);
            atomicAdd(&gsq2[tid],  sStat[C2 + tid]);
        }
    } else {
        // zero pad rows of h2 panel (per-wave; DS ops are in-order within a wave)
        for (int i = lane; i < C2; i += 32) {
            sH2[w][i] = (_Float16)0.0f;
            sH2[w][33 * C2 + i] = (_Float16)0.0f;
        }
        // conv3 + tanh; lane = t
        const int t = lane;
        float o = b3[0];
        for (int co = 0; co < C2; ++co) {
            float wm = w3[co * 3 + 2], w0 = w3[co * 3 + 1], wp = w3[co * 3 + 0];
            o = fmaf((float)sH2[w][t * C2 + co], wm, o);
            o = fmaf((float)sH2[w][(t + 1) * C2 + co], w0, o);
            o = fmaf((float)sH2[w][(t + 2) * C2 + co], wp, o);
        }
        out[(size_t)b * TLEN + t] = tanhf(o);
    }
}

// ---------------------------------------------------------------------------
extern "C" void kernel_launch(void* const* d_in, const int* in_sizes, int n_in,
                              void* d_out, int out_size, void* d_ws, size_t ws_size,
                              hipStream_t stream)
{
    const float* x   = (const float*)d_in[0];
    const float* qp  = (const float*)d_in[1];
    const float* w1  = (const float*)d_in[2];
    const float* b1  = (const float*)d_in[3];
    const float* g1  = (const float*)d_in[4];
    const float* be1 = (const float*)d_in[5];
    const float* w2  = (const float*)d_in[6];
    const float* b2  = (const float*)d_in[7];
    const float* g2  = (const float*)d_in[8];
    const float* be2 = (const float*)d_in[9];
    const float* w3  = (const float*)d_in[10];
    const float* b3  = (const float*)d_in[11];
    float* out = (float*)d_out;

    const int B = in_sizes[0] / 4;          // 16384

    float* images = (float*)d_ws;
    float* stat   = images + (size_t)B * 32;
    float* sum1 = stat;        float* sq1 = stat + 128;
    float* sc1  = stat + 256;  float* sh1 = stat + 384;
    float* sum2 = stat + 512;  float* sq2 = stat + 576;
    float* sc2  = stat + 640;  float* sh2 = stat + 704;
    _Float16* Bp = (_Float16*)(stat + 768);

    const float invN = 1.0f / ((float)B * (float)TLEN);

    k_init<<<1, 256, 0, stream>>>(w2, stat, Bp);
    k_patches<<<(B + 255) / 256, 256, 0, stream>>>(x, qp, images, B);
    k_stats1<<<dim3(8, 16), 256, 0, stream>>>(images, w1, b1, sum1, sq1, B / 16);
    k_finalize<<<1, 128, 0, stream>>>(sum1, sq1, g1, be1, sc1, sh1, 128, invN);
    k_conv2<0><<<B / 4, 128, 0, stream>>>(images, w1, b1, sc1, sh1, Bp, b2,
                                          nullptr, nullptr, nullptr, nullptr,
                                          sum2, sq2, nullptr);
    k_finalize<<<1, 64, 0, stream>>>(sum2, sq2, g2, be2, sc2, sh2, 64, invN);
    k_conv2<1><<<B / 4, 128, 0, stream>>>(images, w1, b1, sc1, sh1, Bp, b2,
                                          sc2, sh2, w3, b3,
                                          nullptr, nullptr, out);
}
